// HeteroRGCN_39522289058325
// MI455X (gfx1250) — matrix operand order repaired and checked by reading, not compiled, not run on hardware
//
#include <hip/hip_runtime.h>
#include <hip/hip_bf16.h>

// ---------------- CDNA5 types ----------------
typedef __bf16 bf16_t;
typedef bf16_t v16bf __attribute__((ext_vector_type(16)));
typedef float  v8f   __attribute__((ext_vector_type(8)));

// TDM descriptor word groups (clang-23 6-arg builtin form:
//   (uint32x4 g0, int32x8 g1, int32x4, int32x4, int32x8, i32 cpol) )
typedef unsigned int tdm_v4u __attribute__((ext_vector_type(4)));
typedef int          tdm_v8i __attribute__((ext_vector_type(8)));
typedef int          tdm_v4i __attribute__((ext_vector_type(4)));

#if defined(__has_builtin)
#if __has_builtin(__builtin_amdgcn_tensor_load_to_lds) && \
    __has_builtin(__builtin_amdgcn_s_wait_tensorcnt)
#define USE_TDM 1
#endif
#endif

#if defined(USE_TDM)
// One 2D TDM tile load: `rows` rows of `width` f32, global row pitch = K
// elements, packed contiguously (width f32 per row) at LDS byte `lds_off`.
// D# bitfields per CDNA5 ISA.
__device__ __forceinline__ void tdm_load_tile(const float* gsrc, unsigned lds_off,
                                              int rows, int width, int K)
{
    const unsigned long long ga = (unsigned long long)(const void*)gsrc;
    const unsigned dim0 = (unsigned)width;     // tile/tensor X extent (elements)
    const unsigned dim1 = (unsigned)rows;      // tile/tensor Y extent (rows)
    const unsigned long long st0 = (unsigned long long)(unsigned)K;  // row pitch

    tdm_v4u g0;
    g0.x = 1u;                                              // count=1, user-mode
    g0.y = lds_off;                                         // lds_addr
    g0.z = (unsigned)(ga & 0xFFFFFFFFull);                  // global_addr[31:0]
    g0.w = (unsigned)((ga >> 32) & 0x1FFFFFFull) | (2u << 30);  // addr[56:32] | type=2

    tdm_v8i g1;
    g1[0] = (int)(2u << 16);                                // data_size = 4B
    g1[1] = (int)((dim0 & 0xFFFFu) << 16);                  // tensor_dim0 (bits 79:48)
    g1[2] = (int)((dim0 >> 16) | ((dim1 & 0xFFFFu) << 16)); // | tensor_dim1 lo
    g1[3] = (int)((dim1 >> 16) | (dim0 << 16));             // | tile_dim0
    g1[4] = (int)dim1;                                      // tile_dim1, tile_dim2=0
    g1[5] = (int)(unsigned)(st0 & 0xFFFFFFFFull);           // tensor_dim0_stride lo
    g1[6] = (int)(unsigned)((st0 >> 32) & 0xFFFFull);       // stride hi; dim1_stride=0
    g1[7] = 0;

    const tdm_v4i gz4 = {0, 0, 0, 0};                       // groups 2/3 (2D tensor)
    const tdm_v8i gz8 = {0, 0, 0, 0, 0, 0, 0, 0};
    __builtin_amdgcn_tensor_load_to_lds(g0, g1, gz4, gz4, gz8, 0);
}
#else
// Fallback: cooperative 256-thread float4 copy of a slab into LDS.
__device__ __forceinline__ void coop_load_tile(float* dst, const float* src,
                                               int rows, int width, int K, int tid)
{
    const int q4 = width >> 2;
    for (int i = tid; i < rows * q4; i += 256) {
        const int r = i / q4;
        const int q = (i - r * q4) << 2;
        const float4 f = *reinterpret_cast<const float4*>(src + (size_t)r * K + q);
        float* d = dst + r * width + q;
        d[0] = f.x; d[1] = f.y; d[2] = f.z; d[3] = f.w;
    }
}
#endif

// =====================================================================
// C[M,N] = A[M,K] @ B[N,K]^T (+bias, +ReLU opt).  M%16==0, N%16==0, K%KW==0.
// Block = 256 threads = 8 waves covering a bm x bn grid of 16x16 C tiles
// (bn = min(tiles_n, 8), bm = 8/bn).  Per K-step both operand slabs
// (A: bm*16 rows x KW, B: bn*16 rows x KW) are staged into LDS by the
// Tensor Data Mover, double-buffered with s_wait_tensorcnt + barrier.
// All operand reads are ds ops; per-lane global traffic is only C stores.
// f32 -> bf16 in registers; KW/32 v_wmma_f32_16x16x32_bf16 per wave per step.
// =====================================================================
template<int KW>
__global__ __launch_bounds__(256) void
gemm_wmma_bf16(const float* __restrict__ A,
               const float* __restrict__ B,
               const float* __restrict__ bias,   // [N] or nullptr
               float* __restrict__ C,
               int M, int N, int K, int relu, int bm, int bn)
{
    __shared__ float smA[2][4 * 16 * KW];        // up to 64 staged A rows
    __shared__ float smB[2][8 * 16 * KW];        // up to 128 staged B rows

    const int tid  = threadIdx.x;
    const int wave = tid >> 5;
    const int lane = tid & 31;
    const int half = lane >> 4;                  // K-half (A/B frags), M-half (C frag)
    const int l16  = lane & 15;                  // row (A) / col (B, C) within tile

    const int tiles_m = M >> 4;
    const int tiles_n = N >> 4;
    const int wm = wave / bn;
    const int wn = wave - wm * bn;
    const int tm = blockIdx.y * bm + wm;
    const int tn = blockIdx.x * bn + wn;
    const bool active = (tm < tiles_m) && (tn < tiles_n);   // wave-uniform

    const int row0  = blockIdx.y * bm * 16;                 // first staged A row
    const int arows = (M - row0 < bm * 16) ? (M - row0) : (bm * 16);
    const int col0  = blockIdx.x * bn * 16;                 // first staged B row
    const int brows = (N - col0 < bn * 16) ? (N - col0) : (bn * 16);

    const float* __restrict__ Ab = A + (size_t)row0 * (size_t)K;
    const float* __restrict__ Bb = B + (size_t)col0 * (size_t)K;

    const int steps = K / KW;
    v8f acc = {};

#if defined(USE_TDM)
    const unsigned ldsA = (unsigned)(size_t)(void*)&smA[0][0];  // flat low 32 = LDS addr
    const unsigned ldsB = (unsigned)(size_t)(void*)&smB[0][0];
    const unsigned abuf_bytes = sizeof(smA[0]);
    const unsigned bbuf_bytes = sizeof(smB[0]);
    if (wave == 0) {
        tdm_load_tile(Ab, ldsA, arows, KW, K);
        tdm_load_tile(Bb, ldsB, brows, KW, K);
    }
#else
    coop_load_tile(&smA[0][0], Ab, arows, KW, K, tid);
    coop_load_tile(&smB[0][0], Bb, brows, KW, K, tid);
    __syncthreads();
#endif

    for (int s = 0; s < steps; ++s) {
        const int buf = s & 1;
#if defined(USE_TDM)
        if (wave == 0)
            __builtin_amdgcn_s_wait_tensorcnt((short)0);    // slabs for `buf` landed
        __syncthreads();                                    // visible to all; prior reads done
        if (wave == 0 && s + 1 < steps) {
            tdm_load_tile(Ab + (size_t)(s + 1) * KW, ldsA + (buf ^ 1) * abuf_bytes,
                          arows, KW, K);
            tdm_load_tile(Bb + (size_t)(s + 1) * KW, ldsB + (buf ^ 1) * bbuf_bytes,
                          brows, KW, K);
        }
#else
        if (s + 1 < steps) {
            coop_load_tile(&smA[buf ^ 1][0], Ab + (size_t)(s + 1) * KW, arows, KW, K, tid);
            coop_load_tile(&smB[buf ^ 1][0], Bb + (size_t)(s + 1) * KW, brows, KW, K, tid);
        }
#endif
        if (active) {
            const int abase = (wm * 16 + l16) * KW;         // A row within slab
            const int bbase = (wn * 16 + l16) * KW;         // B row within slab
            #pragma unroll
            for (int c = 0; c < KW / 32; ++c) {             // KW=64 -> 2 WMMAs/step
                // A fragment (ds reads): element j of vgpr v holds
                // K = 16*half + 2*v + (j&1)   (contiguous, matches B)
                v16bf a;
                #pragma unroll
                for (int v = 0; v < 8; ++v) {
                    const int kk = abase + c * 32 + (half << 4) + (v << 1);
                    a[2 * v]     = (bf16_t)smA[buf][kk];
                    a[2 * v + 1] = (bf16_t)smA[buf][kk + 1];
                }
                // B fragment: element j holds K = 16*half + j (contiguous 16 f32)
                v16bf b;
                #pragma unroll
                for (int j = 0; j < 16; ++j)
                    b[j] = (bf16_t)smB[buf][bbase + c * 32 + (half << 4) + j];

                acc = __builtin_amdgcn_wmma_f32_16x16x32_bf16(
                          /*neg_a=*/false, a, /*neg_b=*/false, b,
                          /*c_mod=*/(short)0, acc, /*reuse_a=*/false, /*reuse_b=*/false);
            }
        }
#if !defined(USE_TDM)
        __syncthreads();
#endif
    }

    if (active) {
        const float bv = bias ? bias[tn * 16 + l16] : 0.0f;
        #pragma unroll
        for (int r = 0; r < 8; ++r) {           // C frag: vgpr r -> M = 8*half + r, N = l16
            float v = acc[r] + bv;
            if (relu) v = fmaxf(v, 0.0f);
            C[(size_t)(tm * 16 + half * 8 + r) * (size_t)N + (size_t)(tn * 16 + l16)] = v;
        }
    }
}

// =====================================================================
// Edge aggregation: sum[dst] += msg[src], cnt[dst] += 1   (D = 32).
// Node accumulators are 1.3 MB -> L2-resident (192 MB L2): atomics stay on-chip.
// =====================================================================
__global__ void edge_agg32(const float* __restrict__ msg,
                           const int* __restrict__ src,
                           const int* __restrict__ dst,
                           float* __restrict__ sum,
                           float* __restrict__ cnt,
                           int E)
{
    const int idx = blockIdx.x * blockDim.x + threadIdx.x;
    const int e = idx >> 5;
    if (e >= E) return;
    const int d = idx & 31;
    const int s = src[e];
    const int t = dst[e];
    atomicAdd(&sum[(size_t)t * 32 + d], msg[(size_t)s * 32 + d]);
    if (d == 0) atomicAdd(&cnt[t], 1.0f);
}

// mean(etype A) + mean(etype B), then ReLU  (matches reference _layer + relu)
__global__ void finalize_mean32(const float* __restrict__ sa, const float* __restrict__ ca,
                                const float* __restrict__ sb, const float* __restrict__ cb,
                                float* __restrict__ out, int n_nodes)
{
    const int idx = blockIdx.x * blockDim.x + threadIdx.x;
    if (idx >= n_nodes * 32) return;
    const int node = idx >> 5;
    const float cav = ca[node], cbv = cb[node];
    const float va = (cav > 0.0f) ? sa[idx] / fmaxf(cav, 1.0f) : 0.0f;
    const float vb = (cbv > 0.0f) ? sb[idx] / fmaxf(cbv, 1.0f) : 0.0f;
    out[idx] = fmaxf(va + vb, 0.0f);
}

__global__ void zero_f32(float* __restrict__ p, long long n)
{
    const long long i = (long long)blockIdx.x * blockDim.x + threadIdx.x;
    if (i < n) p[i] = 0.0f;
}

// =====================================================================
// Host-side launch
// =====================================================================
static inline int cdiv(long long a, long long b) { return (int)((a + b - 1) / b); }

static void launch_gemm(const float* A, const float* B, const float* bias, float* C,
                        int M, int N, int K, int relu, hipStream_t s)
{
    const int tiles_m = M >> 4;
    const int tiles_n = N >> 4;
    const int bn = (tiles_n >= 8) ? 8 : tiles_n;   // tiles_n in {2, 4, 625}
    const int bm = 8 / bn;
    dim3 grid(cdiv(tiles_n, bn), cdiv(tiles_m, bm));
    if ((K & 63) == 0)   // K in {2048, 1024, 64}: 64-wide slabs, 2 WMMAs/step
        gemm_wmma_bf16<64><<<grid, 256, 0, s>>>(A, B, bias, C, M, N, K, relu, bm, bn);
    else                 // K == 32: single WMMA per tile (score / layer-2 GEMMs)
        gemm_wmma_bf16<32><<<grid, 256, 0, s>>>(A, B, bias, C, M, N, K, relu, bm, bn);
}

extern "C" void kernel_launch(void* const* d_in, const int* in_sizes, int n_in,
                              void* d_out, int out_size, void* d_ws, size_t ws_size,
                              hipStream_t stream)
{
    constexpr int N_NODE = 10000;   // n_drug == n_dis
    constexpr int E      = 320000;
    constexpr int H1     = 64;
    constexpr int H2     = 32;

    // ---- inputs (setup_inputs order) ----
    const float* feat_drug = (const float*)d_in[0];
    const float* feat_dis  = (const float*)d_in[1];
    const int* esrc[4] = { (const int*)d_in[2], (const int*)d_in[4],
                           (const int*)d_in[6], (const int*)d_in[8] };  // dd, ds, sd, ss
    const int* edst[4] = { (const int*)d_in[3], (const int*)d_in[5],
                           (const int*)d_in[7], (const int*)d_in[9] };
    const float* W_drug_emb = (const float*)d_in[10];
    const float* W_dis_emb  = (const float*)d_in[11];
    const float* W_final    = (const float*)d_in[12];
    const float* W1[4] = { (const float*)d_in[13], (const float*)d_in[15],
                           (const float*)d_in[17], (const float*)d_in[19] };
    const float* b1[4] = { (const float*)d_in[14], (const float*)d_in[16],
                           (const float*)d_in[18], (const float*)d_in[20] };
    const float* W2[4] = { (const float*)d_in[21], (const float*)d_in[23],
                           (const float*)d_in[25], (const float*)d_in[27] };
    const float* b2[4] = { (const float*)d_in[22], (const float*)d_in[24],
                           (const float*)d_in[26], (const float*)d_in[28] };

    float* out = (float*)d_out;     // [10000, 10000] f32

    // ---- workspace layout (floats) ----
    float* ws = (float*)d_ws;
    size_t off = 0;
    float* hd0 = ws + off; off += (size_t)N_NODE * H1;
    float* hs0 = ws + off; off += (size_t)N_NODE * H1;
    float* Wh[4];
    for (int i = 0; i < 4; ++i) { Wh[i] = ws + off; off += (size_t)N_NODE * H2; }
    float* sum_[4];
    for (int i = 0; i < 4; ++i) { sum_[i] = ws + off; off += (size_t)N_NODE * H2; }
    float* zero_base = sum_[0];
    float* cnt_[4];
    for (int i = 0; i < 4; ++i) { cnt_[i] = ws + off; off += 10240; }
    const long long zero_span = (long long)(ws + off - zero_base);  // sums + counts
    float* hd1 = ws + off; off += (size_t)N_NODE * H2;
    float* hs1 = ws + off; off += (size_t)N_NODE * H2;
    float* hd2 = ws + off; off += (size_t)N_NODE * H2;
    float* hs2 = ws + off; off += (size_t)N_NODE * H2;
    float* hdf = ws + off; off += (size_t)N_NODE * H2;
    (void)ws_size; (void)n_in; (void)in_sizes; (void)out_size;

    const int agg_threads = E * 32;

    // ---- type-specific input embeddings (HBM-bound: 123 MB of features) ----
    launch_gemm(feat_drug, W_drug_emb, nullptr, hd0, N_NODE, H1, 2048, 0, stream);
    launch_gemm(feat_dis,  W_dis_emb,  nullptr, hs0, N_NODE, H1, 1024, 0, stream);

    // ---- layer 1: per-etype linears (K=64), aggregate, mean+sum+ReLU ----
    launch_gemm(hd0, W1[0], b1[0], Wh[0], N_NODE, H2, H1, 0, stream);  // dd (src drug)
    launch_gemm(hd0, W1[1], b1[1], Wh[1], N_NODE, H2, H1, 0, stream);  // ds (src drug)
    launch_gemm(hs0, W1[2], b1[2], Wh[2], N_NODE, H2, H1, 0, stream);  // sd (src dis)
    launch_gemm(hs0, W1[3], b1[3], Wh[3], N_NODE, H2, H1, 0, stream);  // ss (src dis)

    zero_f32<<<cdiv(zero_span, 256), 256, 0, stream>>>(zero_base, zero_span);
    for (int i = 0; i < 4; ++i)
        edge_agg32<<<cdiv(agg_threads, 256), 256, 0, stream>>>(
            Wh[i], esrc[i], edst[i], sum_[i], cnt_[i], E);
    finalize_mean32<<<cdiv((long long)N_NODE * 32, 256), 256, 0, stream>>>(
        sum_[0], cnt_[0], sum_[2], cnt_[2], hd1, N_NODE);
    finalize_mean32<<<cdiv((long long)N_NODE * 32, 256), 256, 0, stream>>>(
        sum_[1], cnt_[1], sum_[3], cnt_[3], hs1, N_NODE);

    // ---- layer 2: per-etype linears (K=32 -> single WMMA per tile) ----
    launch_gemm(hd1, W2[0], b2[0], Wh[0], N_NODE, H2, H2, 0, stream);
    launch_gemm(hd1, W2[1], b2[1], Wh[1], N_NODE, H2, H2, 0, stream);
    launch_gemm(hs1, W2[2], b2[2], Wh[2], N_NODE, H2, H2, 0, stream);
    launch_gemm(hs1, W2[3], b2[3], Wh[3], N_NODE, H2, H2, 0, stream);

    zero_f32<<<cdiv(zero_span, 256), 256, 0, stream>>>(zero_base, zero_span);
    for (int i = 0; i < 4; ++i)
        edge_agg32<<<cdiv(agg_threads, 256), 256, 0, stream>>>(
            Wh[i], esrc[i], edst[i], sum_[i], cnt_[i], E);
    finalize_mean32<<<cdiv((long long)N_NODE * 32, 256), 256, 0, stream>>>(
        sum_[0], cnt_[0], sum_[2], cnt_[2], hd2, N_NODE);
    finalize_mean32<<<cdiv((long long)N_NODE * 32, 256), 256, 0, stream>>>(
        sum_[1], cnt_[1], sum_[3], cnt_[3], hs2, N_NODE);

    // ---- final projection + 10000x10000 score matrix (store-bound: 400 MB) ----
    launch_gemm(hd2, W_final, nullptr, hdf, N_NODE, H2, H2, 0, stream);
    launch_gemm(hdf, hs2, nullptr, out, N_NODE, N_NODE, H2, 0, stream);
}
